// DGCNN_3702261809320
// MI455X (gfx1250) — compile-verified, matrix operand order checked by reading, not run on hardware
//
#include <hip/hip_runtime.h>
#include <hip/hip_bf16.h>
#include <stdint.h>

#define KNN_K 20
#define NPTS  1024
#define BATCH 8

typedef __attribute__((ext_vector_type(16))) __bf16         v16bf;
typedef __attribute__((ext_vector_type(16))) unsigned short v16u;
typedef __attribute__((ext_vector_type(8)))  float          v8f;

// ---------------- helpers ----------------
__device__ __forceinline__ unsigned short f32_to_bf16(float f) {
  unsigned int u = __float_as_uint(f);
  u += 0x7FFFu + ((u >> 16) & 1u);          // round to nearest even
  return (unsigned short)(u >> 16);
}

// order-preserving float encoding so atomicMax(u32) == float max
__device__ __forceinline__ void atomicMaxFloatEnc(unsigned int* a, float v) {
  unsigned int u = __float_as_uint(v);
  unsigned int e = (u & 0x80000000u) ? ~u : (u | 0x80000000u);
  atomicMax(a, e);
}
__device__ __forceinline__ float decodeMaxEnc(unsigned int e) {
  unsigned int u = (e & 0x80000000u) ? (e & 0x7FFFFFFFu) : ~e;
  return __uint_as_float(u);
}

// ---------------- weight prep: f32 -> zero-padded bf16 (O x 2*Cp) ----------------
__global__ void prep_w_kernel(const float* __restrict__ w, unsigned short* __restrict__ dst,
                              int Cin, int Cp, int Cout) {
  int i = blockIdx.x * blockDim.x + threadIdx.x;
  int tot = Cout * 2 * Cp;
  if (i >= tot) return;
  int o  = i / (2 * Cp);
  int c2 = i - o * 2 * Cp;
  int h  = c2 / Cp;           // 0 = diff half, 1 = ctr half
  int cc = c2 - h * Cp;
  float v = (cc < Cin) ? w[o * (2 * Cin) + h * Cin + cc] : 0.0f;
  dst[i] = f32_to_bf16(v);
}

// ---------------- embedding gather -> point-major (B,N,64) with zero pad ----------------
__global__ void embed_kernel(const int* __restrict__ x, const float* __restrict__ emb,
                             float* __restrict__ h0t) {
  int gid = blockIdx.x * blockDim.x + threadIdx.x;   // B*N*64
  int c  = gid & 63;
  int bn = gid >> 6;
  int xi = x[bn];
  h0t[gid] = (c < 50) ? emb[xi * 50 + c] : 0.0f;
}

// ---------------- per-point squared norms ----------------
__global__ void norms_kernel(const float* __restrict__ xt, float* __restrict__ xx,
                             int RS, int COFF, int CP) {
  int i = blockIdx.x * blockDim.x + threadIdx.x;     // B*N
  const float* row = xt + (size_t)i * RS + COFF;
  float s = 0.f;
  for (int c = 0; c < CP; c += 4) {
    float4 v = *(const float4*)(row + c);
    s += v.x * v.x + v.y * v.y + v.z * v.z + v.w * v.w;
  }
  xx[i] = s;
}

// ---------------- KNN: one block per (b,n); 4-row register tiling; top-k via argmax ----------------
__global__ void __launch_bounds__(256) knn_kernel(const float* __restrict__ xt,
                                                  const float* __restrict__ xx,
                                                  int* __restrict__ idx,
                                                  int RS, int COFF, int CP) {
  __shared__ float ctr[512];
  __shared__ float pd[NPTS];
  __shared__ float rv[256];
  __shared__ int   ri[256];
  int bn = blockIdx.x;
  int b = bn >> 10;
  int t = threadIdx.x;
  const float* crow = xt + (size_t)bn * RS + COFF;
  for (int c = t; c < CP; c += 256) ctr[c] = crow[c];
  __syncthreads();
  float xxn = xx[bn];
  {
    const float* r0 = xt + (size_t)(b * NPTS + t) * RS + COFF;
    const float* r1 = r0 + (size_t)256 * RS;
    const float* r2 = r1 + (size_t)256 * RS;
    const float* r3 = r2 + (size_t)256 * RS;
    float a0 = 0.f, a1 = 0.f, a2 = 0.f, a3 = 0.f;
    for (int c = 0; c < CP; c += 4) {
      float4 cv = *(const float4*)(ctr + c);
      float4 v0 = *(const float4*)(r0 + c);
      float4 v1 = *(const float4*)(r1 + c);
      float4 v2 = *(const float4*)(r2 + c);
      float4 v3 = *(const float4*)(r3 + c);
      a0 += v0.x * cv.x + v0.y * cv.y + v0.z * cv.z + v0.w * cv.w;
      a1 += v1.x * cv.x + v1.y * cv.y + v1.z * cv.z + v1.w * cv.w;
      a2 += v2.x * cv.x + v2.y * cv.y + v2.z * cv.z + v2.w * cv.w;
      a3 += v3.x * cv.x + v3.y * cv.y + v3.z * cv.z + v3.w * cv.w;
    }
    pd[t]       = 2.f * a0 - xxn - xx[b * NPTS + t];
    pd[t + 256] = 2.f * a1 - xxn - xx[b * NPTS + t + 256];
    pd[t + 512] = 2.f * a2 - xxn - xx[b * NPTS + t + 512];
    pd[t + 768] = 2.f * a3 - xxn - xx[b * NPTS + t + 768];
  }
  __syncthreads();
  for (int kk = 0; kk < KNN_K; ++kk) {
    float bv = -3.4e38f; int bi = 0;
    for (int m = t; m < NPTS; m += 256) {
      float v = pd[m];
      if (v > bv || (v == bv && m < bi)) { bv = v; bi = m; }
    }
    rv[t] = bv; ri[t] = bi;
    __syncthreads();
    for (int s = 128; s > 0; s >>= 1) {
      if (t < s) {
        float ov = rv[t + s]; int oi = ri[t + s];
        if (ov > rv[t] || (ov == rv[t] && oi < ri[t])) { rv[t] = ov; ri[t] = oi; }
      }
      __syncthreads();
    }
    if (t == 0) { idx[bn * KNN_K + kk] = ri[0]; pd[ri[0]] = -3.4e38f; }
    __syncthreads();
  }
}

// ---------------- EdgeConv epilogue for one 16x16 accumulator tile ----------------
__device__ __forceinline__ void edge_epilogue(v8f acc, int o_first, int b, int n,
                                              int Cout, int Mrow,
                                              unsigned int* __restrict__ maxenc,
                                              float* __restrict__ stats) {
#pragma unroll
  for (int r = 0; r < 8; ++r) {
    float y = acc[r];
    int o = o_first + r;
    atomicMaxFloatEnc(&maxenc[((size_t)b * Cout + o) * NPTS + n], y);
    float s = y, s2 = y * y;
#pragma unroll
    for (int off = 1; off < 16; off <<= 1) {
      s  += __shfl_xor(s,  off, 16);
      s2 += __shfl_xor(s2, off, 16);
    }
    if (Mrow == 0) {
      atomicAdd(&stats[o], s);
      atomicAdd(&stats[Cout + o], s2);
    }
  }
}

// ---------------- fused EdgeConv: gather -> bf16 WMMA GEMM -> k-max + BN stats ----------------
// grid = (B*N*K/16, Cout/256), block = 256 (8 waves).
// Each wave owns TWO 16-channel A tiles (32 channels); all waves share one 16-column
// B tile per 32-deep K chunk, ping-pong staged in LDS (single barrier per chunk).
__global__ void __launch_bounds__(256) edgeconv_wmma(
    const float* __restrict__ xt, int RS, int COFF, int CP,
    const unsigned short* __restrict__ Wb, int Cout,
    const int* __restrict__ idx,
    unsigned int* __restrict__ maxenc,
    float* __restrict__ stats) {
  __shared__ unsigned short Bt[2][32][18];          // padded: conflict-free strided stores
  __shared__ int mB[16], mN[16], mJ[16];
  int t    = threadIdx.x;
  int lane = t & 31;
  int wave = t >> 5;
  if (t < 16) {
    int gcol = blockIdx.x * 16 + t;                 // flattened (b,n,k) column
    int b = gcol / (NPTS * KNN_K);
    int r = gcol - b * NPTS * KNN_K;
    int n = r / KNN_K;
    int k = r - n * KNN_K;
    mB[t] = b; mN[t] = n;
    mJ[t] = idx[(b * NPTS + n) * KNN_K + k];
  }
  __syncthreads();

  int o0    = blockIdx.y * 256 + wave * 32;         // 2 tiles: o0, o0+16
  int Mrow  = lane & 15;                            // A row (M) == B column (N)
  int kbase = (lane < 16) ? 0 : 8;                  // bf16 16x16x32 lane->K mapping
  const unsigned short* wrow0 = Wb + (size_t)(o0 + Mrow) * (2 * CP);
  const unsigned short* wrow1 = wrow0 + (size_t)16 * (2 * CP);

  // coalesced gather of one 32-K x 16-col B chunk into LDS buffer `buf`
  // mapping: e in [0,512): col = e>>5, k = e&31  -> 32 consecutive lanes read
  // 32 consecutive channels of one row (128B bursts from L2)
  auto stage = [&](int buf, int cb) {
#pragma unroll
    for (int h = 0; h < 2; ++h) {
      int e   = t + h * 256;
      int col = e >> 5;
      int k   = e & 31;
      int c   = cb + k;
      int b = mB[col], n = mN[col];
      float v;
      if (c < CP) {                                 // diff half: x[j]-x[n]
        int j = mJ[col];
        const float* nr = xt + (size_t)(b * NPTS + j) * RS + COFF;
        v = nr[c] - xt[(size_t)(b * NPTS + n) * RS + COFF + c];
        if (k == 0 && c + 32 < CP) __builtin_prefetch(nr + c + 32, 0, 3);
      } else {                                      // ctr half: x[n]
        v = xt[(size_t)(b * NPTS + n) * RS + COFF + (c - CP)];
      }
      Bt[buf][k][col] = f32_to_bf16(v);
    }
  };

  v8f acc0 = {};
  v8f acc1 = {};
  int C2  = 2 * CP;
  int nch = C2 >> 5;
  stage(0, 0);
  for (int ci = 0; ci < nch; ++ci) {
    __syncthreads();                                // staging of `cur` complete
    int cur = ci & 1;
    int cb  = ci << 5;

    v16u ar0, ar1, br;
    const unsigned short* ap0 = wrow0 + cb + kbase;
    const unsigned short* ap1 = wrow1 + cb + kbase;
#pragma unroll
    for (int j = 0; j < 8; ++j) { ar0[j] = ap0[j];      ar1[j] = ap1[j]; }
#pragma unroll
    for (int j = 0; j < 8; ++j) { ar0[8 + j] = ap0[16 + j]; ar1[8 + j] = ap1[16 + j]; }
#pragma unroll
    for (int j = 0; j < 8; ++j) br[j]     = Bt[cur][kbase + j][Mrow];
#pragma unroll
    for (int j = 0; j < 8; ++j) br[8 + j] = Bt[cur][kbase + 16 + j][Mrow];
    if (ci + 1 < nch) {
      __builtin_prefetch(ap0 + 32, 0, 3);           // next chunk's weights
      __builtin_prefetch(ap1 + 32, 0, 3);
      stage(1 - cur, cb + 32);                      // overlap gather with WMMA
    }
    v16bf bv = __builtin_bit_cast(v16bf, br);
    acc0 = __builtin_amdgcn_wmma_f32_16x16x32_bf16(false, __builtin_bit_cast(v16bf, ar0),
                                                   false, bv, (short)0, acc0, false, false);
    acc1 = __builtin_amdgcn_wmma_f32_16x16x32_bf16(false, __builtin_bit_cast(v16bf, ar1),
                                                   false, bv, (short)0, acc1, false, false);
  }

  // epilogue: atomic k-max per (b,o,n) + BN sum/sumsq per channel
  int b = mB[Mrow], n = mN[Mrow];
  int mbase = (lane < 16) ? 0 : 8;
  edge_epilogue(acc0, o0 + mbase,      b, n, Cout, Mrow, maxenc, stats);
  edge_epilogue(acc1, o0 + 16 + mbase, b, n, Cout, Mrow, maxenc, stats);
}

// ---------------- BN + LeakyReLU applied to k-max, scatter into concat buffer ----------------
__global__ void bn_apply_kernel(const unsigned int* __restrict__ maxenc,
                                const float* __restrict__ stats,
                                const float* __restrict__ g, const float* __restrict__ bb,
                                float* __restrict__ xct, int Cout, int coff, float invCnt) {
  int gid = blockIdx.x * blockDim.x + threadIdx.x;  // B*Cout*N
  int n = gid & (NPTS - 1);
  int o = (gid >> 10) % Cout;
  int b = gid / (NPTS * Cout);
  float v = decodeMaxEnc(maxenc[gid]);
  float mean = stats[o] * invCnt;
  float var  = stats[Cout + o] * invCnt - mean * mean;
  float y = (v - mean) * rsqrtf(var + 1e-5f) * g[o] + bb[o];
  y = (y < 0.f) ? 0.2f * y : y;
  xct[(size_t)(b * NPTS + n) * 1536 + coff + o] = y;
}

// ---------------- layer5: 1536 -> 50 conv + BN stats ----------------
__global__ void conv5_kernel(const float* __restrict__ xct, const float* __restrict__ w5,
                             float* __restrict__ y5, float* __restrict__ s5) {
  int gid = blockIdx.x * blockDim.x + threadIdx.x;  // B*50*N
  int n  = gid & (NPTS - 1);
  int bo = gid >> 10;
  int o  = bo % 50;
  int b  = bo / 50;
  const float* xr = xct + (size_t)(b * NPTS + n) * 1536;
  const float* wr = w5 + o * 1536;
  float s = 0.f;
  for (int c = 0; c < 1536; c += 4) {
    float4 xv = *(const float4*)(xr + c);
    float4 wv = *(const float4*)(wr + c);
    s += xv.x * wv.x + xv.y * wv.y + xv.z * wv.z + xv.w * wv.w;
  }
  y5[gid] = s;
  float ss = s, s2 = s * s;
#pragma unroll
  for (int off = 16; off > 0; off >>= 1) { ss += __shfl_xor(ss, off); s2 += __shfl_xor(s2, off); }
  if ((threadIdx.x & 31) == 0) { atomicAdd(&s5[o], ss); atomicAdd(&s5[50 + o], s2); }
}

// ---------------- BN+LReLU + global max/mean pooling -> p (8 x 100) ----------------
__global__ void pool5_kernel(const float* __restrict__ y5, const float* __restrict__ s5,
                             const float* __restrict__ g5, const float* __restrict__ b5,
                             float* __restrict__ p) {
  int t = blockIdx.x * blockDim.x + threadIdx.x;
  if (t >= BATCH * 50) return;
  int b = t / 50, o = t - b * 50;
  float mean = s5[o] * (1.f / 8192.f);
  float var  = s5[50 + o] * (1.f / 8192.f) - mean * mean;
  float inv  = rsqrtf(var + 1e-5f) * g5[o];
  float bias = b5[o];
  const float* row = y5 + (size_t)(b * 50 + o) * NPTS;
  float mx = -3.4e38f, sm = 0.f;
  for (int n = 0; n < NPTS; ++n) {
    float y = (row[n] - mean) * inv + bias;
    y = (y < 0.f) ? 0.2f * y : y;
    mx = fmaxf(mx, y);
    sm += y;
  }
  p[b * 100 + o]      = mx;
  p[b * 100 + 50 + o] = sm * (1.f / (float)NPTS);
}

// ---------------- MLP head (batch of 8) in one block ----------------
__global__ void __launch_bounds__(256) head_kernel(
    const float* __restrict__ p,
    const float* __restrict__ lw1, const float* __restrict__ g6, const float* __restrict__ b6,
    const float* __restrict__ lw2, const float* __restrict__ lb2,
    const float* __restrict__ g7, const float* __restrict__ b7,
    const float* __restrict__ lw3, const float* __restrict__ lb3,
    float* __restrict__ out) {
  __shared__ float P[800];
  __shared__ float Z1[8 * 256];
  __shared__ float Z2[8 * 128];
  int t = threadIdx.x;
  for (int i = t; i < 800; i += 256) P[i] = p[i];
  __syncthreads();
  {
    const float* wr = lw1 + t * 100;
    float z[8];
#pragma unroll
    for (int bi = 0; bi < 8; ++bi) {
      float s = 0.f;
      for (int c = 0; c < 100; ++c) s += wr[c] * P[bi * 100 + c];
      z[bi] = s;
    }
    float mean = 0.f;
    for (int bi = 0; bi < 8; ++bi) mean += z[bi];
    mean *= 0.125f;
    float var = 0.f;
    for (int bi = 0; bi < 8; ++bi) { float d = z[bi] - mean; var += d * d; }
    var *= 0.125f;
    float inv = rsqrtf(var + 1e-5f) * g6[t];
    float bias = b6[t];
    for (int bi = 0; bi < 8; ++bi) {
      float y = (z[bi] - mean) * inv + bias;
      Z1[bi * 256 + t] = (y < 0.f) ? 0.2f * y : y;
    }
  }
  __syncthreads();
  if (t < 128) {
    const float* wr = lw2 + t * 256;
    float z[8];
#pragma unroll
    for (int bi = 0; bi < 8; ++bi) {
      float s = lb2[t];
      for (int c = 0; c < 256; ++c) s += wr[c] * Z1[bi * 256 + c];
      z[bi] = s;
    }
    float mean = 0.f;
    for (int bi = 0; bi < 8; ++bi) mean += z[bi];
    mean *= 0.125f;
    float var = 0.f;
    for (int bi = 0; bi < 8; ++bi) { float d = z[bi] - mean; var += d * d; }
    var *= 0.125f;
    float inv = rsqrtf(var + 1e-5f) * g7[t];
    float bias = b7[t];
    for (int bi = 0; bi < 8; ++bi) {
      float y = (z[bi] - mean) * inv + bias;
      Z2[bi * 128 + t] = (y < 0.f) ? 0.2f * y : y;
    }
  }
  __syncthreads();
  if (t < 80) {
    int bi = t / 10, o = t - bi * 10;
    float s = lb3[o];
    const float* wr = lw3 + o * 128;
    for (int c = 0; c < 128; ++c) s += wr[c] * Z2[bi * 128 + c];
    out[t] = s;
  }
}

// ---------------- workspace layout (bytes, 256-aligned) ----------------
#define OFF_W1B   0u          // 256*128 bf16      = 65536
#define OFF_W2B   65536u      // 256*512 bf16      = 262144
#define OFF_W3B   327680u     // 512*512 bf16      = 524288
#define OFF_W4B   851968u     // 512*1024 bf16     = 1048576
#define OFF_H0T   1900544u    // 8*1024*64 f32     = 2097152
#define OFF_XCT   3997696u    // 8*1024*1536 f32   = 50331648
#define OFF_XX    54329344u   // 8192 f32
#define OFF_IDX   54362112u   // 8*1024*20 i32
#define OFF_MAXE  55017472u   // 8*512*1024 u32    = 16777216
#define OFF_STATS 71794688u   // 1024 f32 (sum, sumsq)
#define OFF_Y5    71798784u   // 8*50*1024 f32     = 1638400
#define OFF_S5    73437184u   // 100 f32
#define OFF_P     73437696u   // 800 f32

extern "C" void kernel_launch(void* const* d_in, const int* in_sizes, int n_in,
                              void* d_out, int out_size, void* d_ws, size_t ws_size,
                              hipStream_t stream) {
  (void)in_sizes; (void)n_in; (void)out_size; (void)ws_size;
  const int*   x   = (const int*)d_in[0];
  const float* emb = (const float*)d_in[1];
  const float* w[4]  = {(const float*)d_in[2],  (const float*)d_in[5],
                        (const float*)d_in[8],  (const float*)d_in[11]};
  const float* g[4]  = {(const float*)d_in[3],  (const float*)d_in[6],
                        (const float*)d_in[9],  (const float*)d_in[12]};
  const float* bb[4] = {(const float*)d_in[4],  (const float*)d_in[7],
                        (const float*)d_in[10], (const float*)d_in[13]};
  const float* w5  = (const float*)d_in[14];
  const float* g5  = (const float*)d_in[15];
  const float* b5  = (const float*)d_in[16];
  const float* lw1 = (const float*)d_in[17];
  const float* g6  = (const float*)d_in[18];
  const float* b6  = (const float*)d_in[19];
  const float* lw2 = (const float*)d_in[20];
  const float* lb2 = (const float*)d_in[21];
  const float* g7  = (const float*)d_in[22];
  const float* b7  = (const float*)d_in[23];
  const float* lw3 = (const float*)d_in[24];
  const float* lb3 = (const float*)d_in[25];

  char* ws = (char*)d_ws;
  unsigned short* WB[4] = {(unsigned short*)(ws + OFF_W1B), (unsigned short*)(ws + OFF_W2B),
                           (unsigned short*)(ws + OFF_W3B), (unsigned short*)(ws + OFF_W4B)};
  float*        h0t    = (float*)(ws + OFF_H0T);
  float*        xct    = (float*)(ws + OFF_XCT);
  float*        xx     = (float*)(ws + OFF_XX);
  int*          idx    = (int*)(ws + OFF_IDX);
  unsigned int* maxenc = (unsigned int*)(ws + OFF_MAXE);
  float*        stats  = (float*)(ws + OFF_STATS);
  float*        y5     = (float*)(ws + OFF_Y5);
  float*        s5     = (float*)(ws + OFF_S5);
  float*        p      = (float*)(ws + OFF_P);

  const int Cin[4]  = {50, 256, 256, 512};
  const int Cp[4]   = {64, 256, 256, 512};
  const int Cout[4] = {256, 256, 512, 512};
  const int RS[4]   = {64, 1536, 1536, 1536};
  const int COFF[4] = {0, 0, 256, 512};
  const int OC[4]   = {0, 256, 512, 1024};
  const float* xin[4] = {h0t, xct, xct, xct};

  for (int l = 0; l < 4; ++l) {
    int tot = Cout[l] * 2 * Cp[l];
    prep_w_kernel<<<(tot + 255) / 256, 256, 0, stream>>>(w[l], WB[l], Cin[l], Cp[l], Cout[l]);
  }
  embed_kernel<<<(BATCH * NPTS * 64) / 256, 256, 0, stream>>>(x, emb, h0t);

  const int colTiles = BATCH * NPTS * KNN_K / 16;   // 10240
  for (int l = 0; l < 4; ++l) {
    norms_kernel<<<(BATCH * NPTS) / 256, 256, 0, stream>>>(xin[l], xx, RS[l], COFF[l], Cp[l]);
    knn_kernel<<<BATCH * NPTS, 256, 0, stream>>>(xin[l], xx, idx, RS[l], COFF[l], Cp[l]);
    hipMemsetAsync(maxenc, 0, (size_t)BATCH * Cout[l] * NPTS * 4, stream);
    hipMemsetAsync(stats, 0, (size_t)Cout[l] * 2 * 4, stream);
    edgeconv_wmma<<<dim3(colTiles, Cout[l] / 256), 256, 0, stream>>>(
        xin[l], RS[l], COFF[l], Cp[l], WB[l], Cout[l], idx, maxenc, stats);
    bn_apply_kernel<<<(BATCH * Cout[l] * NPTS) / 256, 256, 0, stream>>>(
        maxenc, stats, g[l], bb[l], xct, Cout[l], OC[l],
        1.0f / (float)(BATCH * NPTS * KNN_K));
  }

  hipMemsetAsync(s5, 0, 512, stream);
  conv5_kernel<<<(BATCH * 50 * NPTS) / 256, 256, 0, stream>>>(xct, w5, y5, s5);
  pool5_kernel<<<1, 512, 0, stream>>>(y5, s5, g5, b5, p);
  head_kernel<<<1, 256, 0, stream>>>(p, lw1, g6, b6, lw2, lb2, g7, b7, lw3, lb3,
                                     (float*)d_out);
}
